// LoFTREncoderLayer_86492051407494
// MI455X (gfx1250) — compile-verified, matrix operand order checked by reading, not run on hardware
//
#include <hip/hip_runtime.h>
#include <hip/hip_bf16.h>

// ---------------------------------------------------------------------------
// LoFTR encoder layer ('self' path) for MI455X / gfx1250, wave32 + WMMA f16.
// b=4, H=120, W=160, L=19200, d=256, h=8, hd=32.
// ---------------------------------------------------------------------------

typedef _Float16 f16;
typedef __attribute__((ext_vector_type(16))) _Float16     v16h;
typedef __attribute__((ext_vector_type(8)))  float        v8f;
typedef __attribute__((ext_vector_type(8)))  unsigned int v8u;

#define BATCH 4
#define IMH   120
#define IMW   160
#define LTOK  (IMH * IMW)          // 19200
#define MROWS (BATCH * LTOK)       // 76800
#define DMODEL 256
#define NHEAD  8
#define HDIM   32

static __device__ __forceinline__ v8f wmma_f16(v16h a, v16h b, v8f c) {
  // D = A(16x32,f16) * B(32x16,f16) + C(16x16,f32)
  return __builtin_amdgcn_wmma_f32_16x16x32_f16(
      /*neg_a=*/false, a, /*neg_b=*/false, b,
      /*c_mod=*/(short)0, c, /*reuse_a=*/false, /*reuse_b=*/false);
}

// k index inside a 16x32 f16 A (or mirrored B) fragment, per CDNA5 ISA layout:
// lanes 0-15 hold k-base 0 (v0..v3 -> k 0..7, v4..v7 -> k 16..23),
// lanes 16-31 hold k-base 8 (k 8..15, 24..31).
static __device__ __forceinline__ int kmap(int e, int kb) {
  return (e < 8) ? (kb + e) : (16 + kb + (e - 8));
}

// ---------------------------------------------------------------------------
// fp32 -> fp16 conversion
// ---------------------------------------------------------------------------
__global__ void cvt_f32_f16(const float* __restrict__ src, f16* __restrict__ dst, long n) {
  long i = (long)blockIdx.x * blockDim.x + threadIdx.x;
  if (i < n) dst[i] = (f16)src[i];
}

// ---------------------------------------------------------------------------
// Generic WMMA GEMM:  C[M,N] = act( [A0 | A1][M,K] @ Bw[K,N] )
//   A0 has K0 columns, A1 (optional) has K-K0 columns (implements concat).
// Block: 128 threads (4 waves). Block tile 128(M) x 64(N), BLOCK_K=32.
// Each wave owns 32 rows x 64 cols -> 2 A-frags x 4 B-frags = 8 WMMA / k-step.
// EPI: 0 = f32 store, 1 = f16 store, 2 = relu->f16, 3 = raw f16 + (elu+1) f16
// ---------------------------------------------------------------------------
template <int EPI>
__device__ __forceinline__ void epi_store(const v8f& cc, int mbase, int nbase, int N,
                                          float* C32, f16* C16, f16* C16b, int lane) {
  int n  = nbase + (lane & 15);
  int mb = mbase + ((lane >> 4) * 8);
#pragma unroll
  for (int r = 0; r < 8; r++) {
    float v = cc[r];
    size_t idx = (size_t)(mb + r) * N + n;
    if (EPI == 0) {
      C32[idx] = v;
    } else if (EPI == 1) {
      C16[idx] = (f16)v;
    } else if (EPI == 2) {
      C16[idx] = (f16)(v > 0.f ? v : 0.f);
    } else {
      C16[idx]  = (f16)v;                                   // raw q/k (conv branch)
      C16b[idx] = (f16)(v > 0.f ? v + 1.f : __expf(v));     // elu(v)+1 feature map
    }
  }
}

template <int EPI>
__global__ __launch_bounds__(128) void gemm_wmma(
    const f16* __restrict__ A0, const f16* __restrict__ A1, int K0, int K,
    const f16* __restrict__ Bw, int N, int M,
    float* __restrict__ C32, f16* __restrict__ C16, f16* __restrict__ C16b) {
  __shared__ __align__(16) f16 As[128 * 40];  // [m][k], pad 8 halfs
  __shared__ __align__(16) f16 Bs[32 * 72];   // [k][n], pad 8 halfs

  const int tid  = threadIdx.x;
  const int n0   = blockIdx.x * 64;
  const int m0   = blockIdx.y * 128;
  const int lane = tid & 31;
  const int wid  = tid >> 5;                  // 0..3
  const int wm   = wid * 32;                  // wave's 32-row slice
  const int mrow = lane & 15;
  const int kb   = (lane >> 4) * 8;
  const int K1   = K - K0;

  v8f acc[2][4];
#pragma unroll
  for (int mi = 0; mi < 2; mi++)
#pragma unroll
    for (int ni = 0; ni < 4; ni++) acc[mi][ni] = (v8f){};

  for (int k0 = 0; k0 < K; k0 += 32) {
    // stage A tile 128x32 (8-half vector loads; concat handled per 8-chunk)
#pragma unroll
    for (int it = 0; it < 4; it++) {
      int idx = tid + it * 128;
      int r  = idx >> 2;
      int c8 = (idx & 3) * 8;
      int gk = k0 + c8;
      const f16* sp;
      if (gk < K0) sp = A0 + (size_t)(m0 + r) * K0 + gk;
      else         sp = A1 + (size_t)(m0 + r) * K1 + (gk - K0);
      *reinterpret_cast<uint4*>(&As[r * 40 + c8]) = *reinterpret_cast<const uint4*>(sp);
    }
    // stage B tile 32x64 (coalesced)
#pragma unroll
    for (int it = 0; it < 2; it++) {
      int idx = tid + it * 128;
      int r  = idx >> 3;
      int n8 = (idx & 7) * 8;
      *reinterpret_cast<uint4*>(&Bs[r * 72 + n8]) =
          *reinterpret_cast<const uint4*>(Bw + (size_t)(k0 + r) * N + n0 + n8);
    }
    __syncthreads();

    // A fragments: element pairs (2p,2p+1) are k-consecutive -> b32 LDS reads
    v8u au0, au1;
    const int row0 = (wm + mrow) * 40;
    const int row1 = (wm + 16 + mrow) * 40;
#pragma unroll
    for (int p = 0; p < 8; p++) {
      int kk = kmap(2 * p, kb);
      au0[p] = *reinterpret_cast<const unsigned int*>(&As[row0 + kk]);
      au1[p] = *reinterpret_cast<const unsigned int*>(&As[row1 + kk]);
    }
    v16h a0 = __builtin_bit_cast(v16h, au0);
    v16h a1 = __builtin_bit_cast(v16h, au1);

    v16h b[4];
#pragma unroll
    for (int ni = 0; ni < 4; ni++) {
#pragma unroll
      for (int e = 0; e < 16; e++) {
        int kk = kmap(e, kb);
        b[ni][e] = Bs[kk * 72 + ni * 16 + mrow];
      }
    }

#pragma unroll
    for (int ni = 0; ni < 4; ni++) {
      acc[0][ni] = wmma_f16(a0, b[ni], acc[0][ni]);
      acc[1][ni] = wmma_f16(a1, b[ni], acc[1][ni]);
    }
    __syncthreads();
  }

#pragma unroll
  for (int mi = 0; mi < 2; mi++)
#pragma unroll
    for (int ni = 0; ni < 4; ni++)
      epi_store<EPI>(acc[mi][ni], m0 + wm + mi * 16, n0 + ni * 16, N,
                     C32, C16, C16b, lane);
}

// ---------------------------------------------------------------------------
// KV[b,h,32,32] = sum_s Kf[b,s,h,:] (x) V[b,s,h,:]   (WMMA over S)
// One wave per (b,h). A[m=dim][k=s] = Kf, B[k=s][n=dim] = V.
// ---------------------------------------------------------------------------
__global__ __launch_bounds__(32) void kv_wmma(const f16* __restrict__ Kf,
                                              const f16* __restrict__ Vh,
                                              float* __restrict__ KV, int S) {
  __shared__ __align__(16) f16 Kt[32 * 40];   // [s][dim]
  __shared__ __align__(16) f16 Vt[32 * 40];
  const int bh   = blockIdx.x;                // b*8 + h
  const int b    = bh >> 3;
  const int h    = bh & 7;
  const int lane = threadIdx.x;
  const int mrow = lane & 15;
  const int kb   = (lane >> 4) * 8;
  const size_t base = (size_t)b * LTOK * DMODEL + (size_t)h * HDIM;

  v8f c00 = {}, c01 = {}, c10 = {}, c11 = {};

  for (int s0 = 0; s0 < S; s0 += 32) {
    size_t off = base + (size_t)(s0 + lane) * DMODEL;
    const uint4* kp = reinterpret_cast<const uint4*>(Kf + off);
    const uint4* vp = reinterpret_cast<const uint4*>(Vh + off);
    uint4* kd = reinterpret_cast<uint4*>(&Kt[lane * 40]);
    uint4* vd = reinterpret_cast<uint4*>(&Vt[lane * 40]);
    kd[0] = kp[0]; kd[1] = kp[1]; kd[2] = kp[2]; kd[3] = kp[3];
    vd[0] = vp[0]; vd[1] = vp[1]; vd[2] = vp[2]; vd[3] = vp[3];
    __syncthreads();

    v16h a0, a1, b0, b1;
#pragma unroll
    for (int e = 0; e < 16; e++) {
      int kk = kmap(e, kb);
      a0[e] = Kt[kk * 40 + mrow];        // A m=dim (transposed read)
      a1[e] = Kt[kk * 40 + 16 + mrow];
      b0[e] = Vt[kk * 40 + mrow];        // B n=dim
      b1[e] = Vt[kk * 40 + 16 + mrow];
    }
    c00 = wmma_f16(a0, b0, c00);
    c01 = wmma_f16(a0, b1, c01);
    c10 = wmma_f16(a1, b0, c10);
    c11 = wmma_f16(a1, b1, c11);
    __syncthreads();
  }

  float* out = KV + (size_t)bh * HDIM * HDIM;
#pragma unroll
  for (int r = 0; r < 8; r++) {
    int m = (lane >> 4) * 8 + r;
    int n = lane & 15;
    out[m * HDIM + n]             = c00[r];
    out[m * HDIM + n + 16]        = c01[r];
    out[(m + 16) * HDIM + n]      = c10[r];
    out[(m + 16) * HDIM + n + 16] = c11[r];
  }
}

// ---------------------------------------------------------------------------
// Ksum[b,h,dim] = sum_s Kf[b,s,h,dim]
// ---------------------------------------------------------------------------
__global__ __launch_bounds__(256) void ksum_kernel(const f16* __restrict__ Kf,
                                                   float* __restrict__ Ksum, int S) {
  const int bh = blockIdx.x;
  const int b  = bh >> 3, h = bh & 7;
  const int t  = threadIdx.x;
  const int dim = t & 31, chunk = t >> 5;   // 8 s-chunks
  const size_t base = (size_t)b * LTOK * DMODEL + (size_t)h * HDIM + dim;
  float acc = 0.f;
  for (int s = chunk; s < S; s += 8) acc += (float)Kf[base + (size_t)s * DMODEL];
  __shared__ float red[256];
  red[t] = acc;
  __syncthreads();
  if (chunk == 0) {
    float v = red[dim];
#pragma unroll
    for (int c = 1; c < 8; c++) v += red[c * 32 + dim];
    Ksum[bh * 32 + dim] = v;
  }
}

// ---------------------------------------------------------------------------
// message[b,l,h,v] = (Qf . KV[:,v]) / (Qf . Ksum + 1e-6)   (1/S and *S cancel)
// ---------------------------------------------------------------------------
__global__ __launch_bounds__(256) void attn_msg(const f16* __restrict__ Qf,
                                                const float* __restrict__ KV,
                                                const float* __restrict__ Ksum,
                                                f16* __restrict__ msgh) {
  const int row = blockIdx.x;          // b*L + l
  const int b   = row / LTOK;
  const int t   = threadIdx.x;
  __shared__ f16 qrow[DMODEL];
  qrow[t] = Qf[(size_t)row * DMODEL + t];
  __syncthreads();
  const int h = t >> 5, vd = t & 31;
  const int bh = b * NHEAD + h;
  const float* kv = KV + (size_t)bh * HDIM * HDIM;
  const float* ks = Ksum + bh * HDIM;
  float z = 0.f, acc = 0.f;
#pragma unroll 8
  for (int dd = 0; dd < HDIM; dd++) {
    float q = (float)qrow[h * HDIM + dd];
    z   += q * ks[dd];
    acc += q * kv[dd * HDIM + vd];
  }
  msgh[(size_t)row * DMODEL + t] = (f16)(acc / (z + 1e-6f));
}

// ---------------------------------------------------------------------------
// LayerNorm over 256 features; optional f32 and f16 outputs.
// ---------------------------------------------------------------------------
__global__ __launch_bounds__(256) void ln_kernel(const float* __restrict__ in,
                                                 const float* __restrict__ w,
                                                 const float* __restrict__ bi,
                                                 float* __restrict__ out32,
                                                 f16* __restrict__ out16) {
  const int row = blockIdx.x, t = threadIdx.x;
  float v = in[(size_t)row * DMODEL + t];
  __shared__ float s1[256], s2[256];
  s1[t] = v;
  s2[t] = v * v;
  __syncthreads();
  for (int st = 128; st > 0; st >>= 1) {
    if (t < st) { s1[t] += s1[t + st]; s2[t] += s2[t + st]; }
    __syncthreads();
  }
  float mu  = s1[0] * (1.f / DMODEL);
  float var = s2[0] * (1.f / DMODEL) - mu * mu;
  float y   = (v - mu) * rsqrtf(var + 1e-5f) * w[t] + bi[t];
  if (out32) out32[(size_t)row * DMODEL + t] = y;
  if (out16) out16[(size_t)row * DMODEL + t] = (f16)y;
}

// ---------------------------------------------------------------------------
// fc9: f9[b, dcoord*9+o, l] = fc_b[o] + sum_{c<24} fc_w[o,c] * f_all[c]
//   f_all[c = 3*head + {0:q,1:k,2:v}] at (b,l,head*32+dcoord)
// ---------------------------------------------------------------------------
__global__ void fc9_kernel(const f16* __restrict__ qh, const f16* __restrict__ kh,
                           const f16* __restrict__ vh, const float* __restrict__ fcw,
                           const float* __restrict__ fcb, f16* __restrict__ f9) {
  long i = (long)blockIdx.x * blockDim.x + threadIdx.x;   // over B*L*32
  if (i >= (long)BATCH * LTOK * HDIM) return;
  const int  dcoord = (int)(i & 31);
  const long bl     = i >> 5;
  const int  b      = (int)(bl / LTOK);
  const int  l      = (int)(bl % LTOK);
  float in[24];
  const size_t base = (size_t)bl * DMODEL + dcoord;
#pragma unroll
  for (int hi = 0; hi < 8; hi++) {
    in[3 * hi + 0] = (float)qh[base + hi * HDIM];
    in[3 * hi + 1] = (float)kh[base + hi * HDIM];
    in[3 * hi + 2] = (float)vh[base + hi * HDIM];
  }
#pragma unroll
  for (int o = 0; o < 9; o++) {
    float acc = fcb[o];
#pragma unroll
    for (int c = 0; c < 24; c++) acc += fcw[o * 24 + c] * in[c];
    f9[((size_t)b * 288 + dcoord * 9 + o) * LTOK + l] = (f16)acc;
  }
}

// ---------------------------------------------------------------------------
// grouped 3x3 conv: out[b,l,o] = dep_b[o] + sum_{i<9} w[o,i,.,.] * f9[(o/8)*9+i]
// ---------------------------------------------------------------------------
__global__ void dw_kernel(const f16* __restrict__ f9, const float* __restrict__ depw,
                          const float* __restrict__ depb, float* __restrict__ conv) {
  long i = (long)blockIdx.x * blockDim.x + threadIdx.x;   // over B*L*256
  if (i >= (long)BATCH * LTOK * DMODEL) return;
  const int  o  = (int)(i & 255);
  const long bl = i >> 8;
  const int  b  = (int)(bl / LTOK);
  const int  l  = (int)(bl % LTOK);
  const int  y  = l / IMW, x = l % IMW;
  const int  g  = o >> 3;
  float acc = depb[o];
#pragma unroll
  for (int ci = 0; ci < 9; ci++) {
    const f16*   ch = f9 + ((size_t)b * 288 + g * 9 + ci) * LTOK;
    const float* wk = depw + (o * 9 + ci) * 9;
#pragma unroll
    for (int ky = 0; ky < 3; ky++) {
      int yy = y + ky - 1;
      if (yy < 0 || yy >= IMH) continue;
#pragma unroll
      for (int kx = 0; kx < 3; kx++) {
        int xx = x + kx - 1;
        if (xx < 0 || xx >= IMW) continue;
        acc += wk[ky * 3 + kx] * (float)ch[yy * IMW + xx];
      }
    }
  }
  conv[(size_t)bl * DMODEL + o] = acc;
}

// ---------------------------------------------------------------------------
// out = x + msg2ln + 0.1*up + rate*conv
// ---------------------------------------------------------------------------
__global__ void final_kernel(const float* __restrict__ x, const float* __restrict__ m2,
                             const float* __restrict__ up, const float* __restrict__ cv,
                             const float* __restrict__ rate, float* __restrict__ out, long n) {
  long i = (long)blockIdx.x * blockDim.x + threadIdx.x;
  if (i < n) out[i] = x[i] + m2[i] + 0.1f * up[i] + rate[0] * cv[i];
}

// ---------------------------------------------------------------------------
// host
// ---------------------------------------------------------------------------
extern "C" void kernel_launch(void* const* d_in, const int* in_sizes, int n_in,
                              void* d_out, int out_size, void* d_ws, size_t ws_size,
                              hipStream_t stream) {
  (void)in_sizes; (void)n_in; (void)out_size; (void)ws_size;

  const float* x      = (const float*)d_in[0];
  const float* source = (const float*)d_in[1];
  const float* Wq     = (const float*)d_in[2];
  const float* Wk     = (const float*)d_in[3];
  const float* Wv     = (const float*)d_in[4];
  const float* Wmerge = (const float*)d_in[5];
  const float* Wmlp1  = (const float*)d_in[6];
  const float* Wmlp2  = (const float*)d_in[7];
  const float* ln1w   = (const float*)d_in[8];
  const float* ln1b   = (const float*)d_in[9];
  const float* ln2w   = (const float*)d_in[10];
  const float* ln2b   = (const float*)d_in[11];
  const float* Wdown  = (const float*)d_in[12];
  const float* Wup    = (const float*)d_in[13];
  const float* fcw    = (const float*)d_in[14];
  const float* fcb    = (const float*)d_in[15];
  const float* depw   = (const float*)d_in[16];
  const float* depb   = (const float*)d_in[17];
  const float* rate   = (const float*)d_in[18];
  float* out = (float*)d_out;

  char* ws = (char*)d_ws;
  size_t off = 0;
  auto take = [&](size_t bytes) -> size_t {
    size_t o = off;
    off += (bytes + 255) & ~(size_t)255;
    return o;
  };
  const size_t M = MROWS;
  const size_t szMD_h  = M * DMODEL * sizeof(f16);
  const size_t szMD_f  = M * DMODEL * sizeof(float);

  const size_t o_wq    = take(256 * 256 * 2);
  const size_t o_wk    = take(256 * 256 * 2);
  const size_t o_wv    = take(256 * 256 * 2);
  const size_t o_wm    = take(256 * 256 * 2);
  const size_t o_w1    = take(512 * 512 * 2);
  const size_t o_w2    = take(512 * 256 * 2);
  const size_t o_wd    = take(512 * 64 * 2);
  const size_t o_wu    = take(64 * 256 * 2);
  const size_t o_xh    = take(szMD_h);
  const size_t o_sh    = take(szMD_h);
  const size_t o_qh    = take(szMD_h);
  const size_t o_kh    = take(szMD_h);
  const size_t o_vh    = take(szMD_h);   // vh + Kf contiguous -> reused as hid (M x 512 f16)
  const size_t o_Kf    = take(szMD_h);
  const size_t o_Qf    = take(szMD_h);
  const size_t o_msg32 = take(szMD_f);   // merge out -> mlp2 out -> up out
  const size_t o_m2ln  = take(szMD_f);
  const size_t o_t64   = take(M * 64 * 2);
  const size_t o_f9    = take((size_t)BATCH * 288 * LTOK * 2);
  const size_t o_conv  = take(szMD_f);
  const size_t o_KV    = take(32 * 32 * 32 * 4);
  const size_t o_Ksum  = take(32 * 32 * 4);

  f16* wqh = (f16*)(ws + o_wq);   f16* wkh = (f16*)(ws + o_wk);
  f16* wvh = (f16*)(ws + o_wv);   f16* wmh = (f16*)(ws + o_wm);
  f16* w1h = (f16*)(ws + o_w1);   f16* w2h = (f16*)(ws + o_w2);
  f16* wdh = (f16*)(ws + o_wd);   f16* wuh = (f16*)(ws + o_wu);
  f16* xh  = (f16*)(ws + o_xh);   f16* sh  = (f16*)(ws + o_sh);
  f16* qh  = (f16*)(ws + o_qh);   f16* kh  = (f16*)(ws + o_kh);
  f16* vh  = (f16*)(ws + o_vh);   f16* Kf  = (f16*)(ws + o_Kf);
  f16* Qf  = (f16*)(ws + o_Qf);
  float* msg32 = (float*)(ws + o_msg32);
  float* m2ln  = (float*)(ws + o_m2ln);
  f16*   t64   = (f16*)(ws + o_t64);
  f16*   f9    = (f16*)(ws + o_f9);
  float* conv  = (float*)(ws + o_conv);
  float* KV    = (float*)(ws + o_KV);
  float* Ksum  = (float*)(ws + o_Ksum);

  f16* msgh  = qh;   // reuse: qh free after fc9
  f16* ln1h  = kh;   // reuse: kh free after fc9
  f16* hid   = vh;   // reuse: vh+Kf span = M x 512 f16
  f16* m2lnh = Qf;   // reuse: Qf free after attn_msg
  float* up32 = msg32;

  auto cvt = [&](const float* s, f16* d, long n) {
    long blocks = (n + 255) / 256;
    cvt_f32_f16<<<dim3((unsigned)blocks), dim3(256), 0, stream>>>(s, d, n);
  };

  // 1) fp16 conversions (weights + activations)
  cvt(Wq, wqh, 256 * 256);   cvt(Wk, wkh, 256 * 256);
  cvt(Wv, wvh, 256 * 256);   cvt(Wmerge, wmh, 256 * 256);
  cvt(Wmlp1, w1h, 512 * 512); cvt(Wmlp2, w2h, 512 * 256);
  cvt(Wdown, wdh, 512 * 64);  cvt(Wup, wuh, 64 * 256);
  cvt(x, xh, (long)M * DMODEL);
  cvt(source, sh, (long)M * DMODEL);

  const dim3 blk128(128);
  auto ggrid = [&](int N) { return dim3(N / 64, (unsigned)(M / 128)); };

  // 2) q/k/v projections (epilogue emits raw f16 + elu+1 feature maps)
  gemm_wmma<3><<<ggrid(256), blk128, 0, stream>>>(xh, nullptr, 256, 256, wqh, 256, (int)M, nullptr, qh, Qf);
  gemm_wmma<3><<<ggrid(256), blk128, 0, stream>>>(sh, nullptr, 256, 256, wkh, 256, (int)M, nullptr, kh, Kf);
  gemm_wmma<1><<<ggrid(256), blk128, 0, stream>>>(sh, nullptr, 256, 256, wvh, 256, (int)M, nullptr, vh, nullptr);

  // 3) conv branch (1x1 fc 24->9, then grouped 3x3)
  {
    long n = (long)BATCH * LTOK * HDIM;
    fc9_kernel<<<dim3((unsigned)((n + 255) / 256)), dim3(256), 0, stream>>>(qh, kh, vh, fcw, fcb, f9);
    long m = (long)BATCH * LTOK * DMODEL;
    dw_kernel<<<dim3((unsigned)((m + 255) / 256)), dim3(256), 0, stream>>>(f9, depw, depb, conv);
  }

  // 4) linear attention: Ksum + KV (WMMA over S)
  ksum_kernel<<<dim3(32), dim3(256), 0, stream>>>(Kf, Ksum, LTOK);
  kv_wmma<<<dim3(32), dim3(32), 0, stream>>>(Kf, vh, KV, LTOK);

  // 5) message = Qf @ KV * Z
  attn_msg<<<dim3((unsigned)M), dim3(256), 0, stream>>>(Qf, KV, Ksum, msgh);

  // 6) merge GEMM + LN1
  gemm_wmma<0><<<ggrid(256), blk128, 0, stream>>>(msgh, nullptr, 256, 256, wmh, 256, (int)M, msg32, nullptr, nullptr);
  ln_kernel<<<dim3((unsigned)M), dim3(256), 0, stream>>>(msg32, ln1w, ln1b, nullptr, ln1h);

  // 7) MLP: relu([x|msg] @ Wmlp1) @ Wmlp2, then LN2
  gemm_wmma<2><<<ggrid(512), blk128, 0, stream>>>(xh, ln1h, 256, 512, w1h, 512, (int)M, nullptr, hid, nullptr);
  gemm_wmma<0><<<ggrid(256), blk128, 0, stream>>>(hid, nullptr, 512, 512, w2h, 256, (int)M, msg32, nullptr, nullptr);
  ln_kernel<<<dim3((unsigned)M), dim3(256), 0, stream>>>(msg32, ln2w, ln2b, m2ln, m2lnh);

  // 8) adapter: up = relu([x|msg2] @ Wdown) @ Wup
  gemm_wmma<2><<<ggrid(64),  blk128, 0, stream>>>(xh, m2lnh, 256, 512, wdh, 64, (int)M, nullptr, t64, nullptr);
  gemm_wmma<0><<<ggrid(256), blk128, 0, stream>>>(t64, nullptr, 64, 64, wuh, 256, (int)M, up32, nullptr, nullptr);

  // 9) final residual combine
  {
    long n = (long)M * DMODEL;
    final_kernel<<<dim3((unsigned)((n + 255) / 256)), dim3(256), 0, stream>>>(
        x, m2ln, up32, conv, rate, out, n);
  }
}
